// SpatialAttention_80083960201678
// MI455X (gfx1250) — compile-verified, matrix operand order checked by reading, not run on hardware
//
#include <hip/hip_runtime.h>
#include <hip/hip_bf16.h>

// ---------------------------------------------------------------------------
// SpatialAttention for MI455X (gfx1250, wave32, WMMA).
// Shapes: B=2, C=256, N=2048 (=16*16*8), heads=8, head_dim=32.
// All matmuls run on v_wmma_f32_16x16x32_f16 (f16 in, f32 accumulate).
// Attention stages K/V tiles through LDS shared by 8 waves per workgroup.
// ---------------------------------------------------------------------------

typedef _Float16 v16h __attribute__((ext_vector_type(16)));
typedef _Float16 v8h  __attribute__((ext_vector_type(8)));
typedef float    v8f  __attribute__((ext_vector_type(8)));

#define BB 2
#define CC 256
#define NN 2048
#define NH 8
#define HD 32

// WMMA 16x16x32 f16 fragment conventions (wave32), per CDNA5 ISA 7.12.2:
//  A (16x32, M x K): lane L holds row M = L%16; element e holds
//      K = (e<8) ? base+e : 16+base+(e-8), base = (L/16)*8.
//  B (32x16, K x N): mirrored — lane L holds column N = L%16, same K map.
//  C/D (16x16 f32):  lane L holds column N = L%16; VGPR r holds row M = r + 8*(L/16).

__device__ __forceinline__ v16h load_frag_f16(const _Float16* p_lo, const _Float16* p_hi) {
    v8h lo = *(const v8h*)p_lo;
    v8h hi = *(const v8h*)p_hi;
    v16h r;
#pragma unroll
    for (int e = 0; e < 8; ++e) { r[e] = lo[e]; r[e + 8] = hi[e]; }
    return r;
}

__device__ __forceinline__ v8f wmma_f16(v16h a, v16h b, v8f c) {
    return __builtin_amdgcn_wmma_f32_16x16x32_f16(false, a, false, b, (short)0, c,
                                                  false, false);
}

// ---------------------------------------------------------------------------
// Kernel 1: x (B,C,N) f32 -> xt (B,N,C) f16, LDS-tiled transpose.
// ---------------------------------------------------------------------------
__global__ __launch_bounds__(256) void k_xpose(const float* __restrict__ x,
                                               _Float16* __restrict__ xt) {
    __shared__ _Float16 tile[16][17];
    const int b  = blockIdx.z;
    const int c0 = blockIdx.y * 16;
    const int n0 = blockIdx.x * 16;
    const int tx = threadIdx.x & 15;   // n within tile on read, c on write
    const int ty = threadIdx.x >> 4;   // c within tile on read, n on write
    tile[ty][tx] = (_Float16)x[((long)b * CC + (c0 + ty)) * NN + (n0 + tx)];
    __syncthreads();
    xt[((long)b * NN + (n0 + ty)) * CC + (c0 + tx)] = tile[tx][ty];
}

// ---------------------------------------------------------------------------
// Kernel 2: QKV GEMM. qkv[o,n] = sum_c w[o,c]*x[b,c,n] + bias[o].
// One wave -> one 16x16 (o x n) tile, K-loop over 256 in steps of 32.
// Scatter epilogue: o = h*96 + s*32 + cl ; q,k -> (b,h,n,32); v -> (b,h,32,n).
// ---------------------------------------------------------------------------
__global__ __launch_bounds__(256) void k_qkv(const _Float16* __restrict__ xt,
                                             const float* __restrict__ w,
                                             const float* __restrict__ bias,
                                             _Float16* __restrict__ q,
                                             _Float16* __restrict__ k,
                                             _Float16* __restrict__ v) {
    const int lane = threadIdx.x & 31;
    const int wid  = threadIdx.x >> 5;
    const int gw   = blockIdx.x * 8 + wid;          // 2*48*128 = 12288 waves
    const int b    = gw / (48 * 128);
    const int rem  = gw % (48 * 128);
    const int ot   = rem / 128;
    const int nt   = rem % 128;
    const int col  = lane & 15;
    const int kb   = (lane >> 4) * 8;               // K pattern base
    const int o0   = ot * 16;
    const int n0   = nt * 16;

    const float*    wrow = w  + (long)(o0 + col) * CC;           // A row
    const _Float16* xrow = xt + ((long)b * NN + n0 + col) * CC;  // B column

    v8f acc = {};
#pragma unroll
    for (int kk = 0; kk < CC; kk += 32) {
        v16h a;
#pragma unroll
        for (int e = 0; e < 8; ++e) {
            a[e]     = (_Float16)wrow[kk + kb + e];
            a[e + 8] = (_Float16)wrow[kk + 16 + kb + e];
        }
        v16h bf = load_frag_f16(xrow + kk + kb, xrow + kk + 16 + kb);
        acc = wmma_f16(a, bf, acc);
    }

    const int n = n0 + col;
#pragma unroll
    for (int r = 0; r < 8; ++r) {
        const int o  = o0 + kb + r;                 // output channel of row r
        const float val = acc[r] + bias[o];
        const int h = o / 96, rm = o % 96, s = rm / 32, cl = rm % 32;
        const _Float16 hv = (_Float16)val;
        if (s == 0)      q[(((long)b * NH + h) * NN + n) * HD + cl] = hv;
        else if (s == 1) k[(((long)b * NH + h) * NN + n) * HD + cl] = hv;
        else             v[(((long)b * NH + h) * HD + cl) * NN + n] = hv;
    }
}

// ---------------------------------------------------------------------------
// Kernel 3: flash attention. One block = 8 waves, all on the same (b,h);
// each wave owns 16 query tokens. 32-key steps: K/V tiles (4 KB) staged
// cooperatively into LDS (shared by all 8 waves -> 8x less L2 traffic),
// online softmax. Computes S^T = K·Q^T so exp(S^T) tiles land directly in
// the PV B-fragment layout (no cross-lane transpose).
// ---------------------------------------------------------------------------
__global__ __launch_bounds__(256) void k_attn(const _Float16* __restrict__ q,
                                              const _Float16* __restrict__ k,
                                              const _Float16* __restrict__ v,
                                              _Float16* __restrict__ aoT) {
    // Rows padded to 40 halves (80 B): keeps b128 LDS ops 16B-aligned and
    // de-strides bank access across the 32 rows.
    __shared__ _Float16 ksh[32][40];   // ksh[m_loc][c]
    __shared__ _Float16 vsh[32][40];   // vsh[c][m_loc]

    const int tid  = threadIdx.x;
    const int lane = tid & 31;
    const int wid  = tid >> 5;
    const int bh   = blockIdx.x >> 4;               // (b,h): 16 blocks each
    const int nt   = (blockIdx.x & 15) * 8 + wid;   // query tile of this wave
    const int col  = lane & 15;
    const int kb   = (lane >> 4) * 8;
    const float scale = 0.17677669529663687f;       // 1/sqrt(32)

    const _Float16* kbase = k + (long)bh * NN * HD;
    const _Float16* vbase = v + (long)bh * HD * NN;

    // Staging role: 256 threads x 16 B == 4 KB (one K tile + one V tile).
    const int srow = (tid & 127) >> 2;              // 0..31
    const int schk = (tid & 3) * 8;                 // half-offset 0,8,16,24
    const bool isK = tid < 128;

    // Q fragment (B operand, loop invariant): row n = nt*16+col, 32 channels.
    const _Float16* qp = q + ((long)bh * NN + nt * 16 + col) * HD;
    const v16h qf = load_frag_f16(qp + kb, qp + 16 + kb);

    v8f o0 = {}, o1 = {};                           // O[c,n]: c 0..15 / 16..31
    float mrun = -1e30f, lrun = 0.0f;

#pragma unroll 1
    for (int mt = 0; mt < NN; mt += 32) {
        // ---- cooperative stage of K/V tiles into LDS ----
        if (isK) {
            *(v8h*)&ksh[srow][schk] =
                *(const v8h*)(kbase + (long)(mt + srow) * HD + schk);
        } else {
            *(v8h*)&vsh[srow][schk] =
                *(const v8h*)(vbase + (long)srow * NN + mt + schk);
        }
        if (mt + 32 < NN) {                         // prefetch next tile
            __builtin_prefetch(
                isK ? (const void*)(kbase + (long)(mt + 32 + srow) * HD + schk)
                    : (const void*)(vbase + (long)srow * NN + mt + 32 + schk),
                0, 3);
        }
        __syncthreads();

        // ---- S^T = K_tile · Q^T : two 16x16 tiles (keys mt..mt+31) ----
        v16h ka0 = load_frag_f16(&ksh[col][kb],      &ksh[col][16 + kb]);
        v16h ka1 = load_frag_f16(&ksh[16 + col][kb], &ksh[16 + col][16 + kb]);
        v8f s0 = {}, s1 = {};
        s0 = wmma_f16(ka0, qf, s0);
        s1 = wmma_f16(ka1, qf, s1);

        // ---- online softmax over m (rows of S^T = 16 regs + partner half) --
        float tmax = -1e30f;
#pragma unroll
        for (int r = 0; r < 8; ++r) {
            s0[r] *= scale; s1[r] *= scale;
            tmax = fmaxf(tmax, fmaxf(s0[r], s1[r]));
        }
        tmax = fmaxf(tmax, __shfl_xor(tmax, 16, 32));
        const float mnew  = fmaxf(mrun, tmax);
        const float alpha = __expf(mrun - mnew);

        float tsum = 0.0f;
        v16h pf;                                    // exp tile, PV B layout
#pragma unroll
        for (int r = 0; r < 8; ++r) {
            const float p0 = __expf(s0[r] - mnew);
            const float p1 = __expf(s1[r] - mnew);
            tsum += p0 + p1;
            pf[r]     = (_Float16)p0;
            pf[r + 8] = (_Float16)p1;
        }
        tsum += __shfl_xor(tsum, 16, 32);
        lrun = lrun * alpha + tsum;
        mrun = mnew;
#pragma unroll
        for (int r = 0; r < 8; ++r) { o0[r] *= alpha; o1[r] *= alpha; }

        // ---- O += V_tile · P  (A: v rows c, K = 32 keys) ----
        v16h va0 = load_frag_f16(&vsh[col][kb],      &vsh[col][16 + kb]);
        v16h va1 = load_frag_f16(&vsh[16 + col][kb], &vsh[16 + col][16 + kb]);
        o0 = wmma_f16(va0, pf, o0);
        o1 = wmma_f16(va1, pf, o1);

        __syncthreads();                            // protect LDS reuse
    }

    // ---- normalize + store token-major: aoT[b, n, h*32 + c] (f16) ----
    const float inv = 1.0f / lrun;
    const int b = bh >> 3, h = bh & 7;
    _Float16* dst = aoT + ((long)b * NN + nt * 16 + col) * CC + h * HD;
    v8h st0, st1;
#pragma unroll
    for (int r = 0; r < 8; ++r) {
        st0[r] = (_Float16)(o0[r] * inv);
        st1[r] = (_Float16)(o1[r] * inv);
    }
    *(v8h*)(dst + kb)      = st0;   // c = kb + r
    *(v8h*)(dst + 16 + kb) = st1;   // c = 16 + kb + r
}

// ---------------------------------------------------------------------------
// Kernel 4: out = x + b_out + w_out · attn_out. fp32 epilogue + residual.
// ---------------------------------------------------------------------------
__global__ __launch_bounds__(256) void k_proj(const _Float16* __restrict__ aoT,
                                              const float* __restrict__ w,
                                              const float* __restrict__ bias,
                                              const float* __restrict__ x,
                                              float* __restrict__ out) {
    const int lane = threadIdx.x & 31;
    const int wid  = threadIdx.x >> 5;
    const int gw   = blockIdx.x * 8 + wid;          // 2*16*128 = 4096 waves
    const int b    = gw / (16 * 128);
    const int rem  = gw % (16 * 128);
    const int ot   = rem / 128;
    const int nt   = rem % 128;
    const int col  = lane & 15;
    const int kb   = (lane >> 4) * 8;
    const int o0   = ot * 16;
    const int n0   = nt * 16;

    const float*    wrow = w   + (long)(o0 + col) * CC;
    const _Float16* arow = aoT + ((long)b * NN + n0 + col) * CC;

    v8f acc = {};
#pragma unroll
    for (int kk = 0; kk < CC; kk += 32) {
        v16h a;
#pragma unroll
        for (int e = 0; e < 8; ++e) {
            a[e]     = (_Float16)wrow[kk + kb + e];
            a[e + 8] = (_Float16)wrow[kk + 16 + kb + e];
        }
        v16h bf = load_frag_f16(arow + kk + kb, arow + kk + 16 + kb);
        acc = wmma_f16(a, bf, acc);
    }

    const int n = n0 + col;
#pragma unroll
    for (int r = 0; r < 8; ++r) {
        const int o = o0 + kb + r;
        const long idx = ((long)b * CC + o) * NN + n;
        out[idx] = x[idx] + acc[r] + bias[o];
    }
}

// ---------------------------------------------------------------------------
// Launch
// ---------------------------------------------------------------------------
extern "C" void kernel_launch(void* const* d_in, const int* in_sizes, int n_in,
                              void* d_out, int out_size, void* d_ws, size_t ws_size,
                              hipStream_t stream) {
    const float* x     = (const float*)d_in[0];   // (2,256,16,16,8)
    const float* w_qkv = (const float*)d_in[1];   // (768,256)
    const float* b_qkv = (const float*)d_in[2];   // (768)
    const float* w_out = (const float*)d_in[3];   // (256,256)
    const float* b_out = (const float*)d_in[4];   // (256)
    float* out = (float*)d_out;                   // (2,256,2048)

    // Workspace partition (f16 buffers, all 256B aligned): 10 MB total.
    char* p = (char*)d_ws;
    _Float16* xt  = (_Float16*)p; p += (size_t)BB * NN * CC * 2;        // (b,n,c)
    _Float16* qb  = (_Float16*)p; p += (size_t)BB * NH * NN * HD * 2;   // (b,h,n,32)
    _Float16* kbf = (_Float16*)p; p += (size_t)BB * NH * NN * HD * 2;   // (b,h,n,32)
    _Float16* vb  = (_Float16*)p; p += (size_t)BB * NH * HD * NN * 2;   // (b,h,32,n)
    _Float16* aoT = (_Float16*)p; p += (size_t)BB * NN * CC * 2;        // (b,n,c)
    (void)n_in; (void)in_sizes; (void)out_size; (void)ws_size;

    k_xpose<<<dim3(NN / 16, CC / 16, BB), 256, 0, stream>>>(x, xt);
    k_qkv  <<<(BB * 48 * 128) / 8, 256, 0, stream>>>(xt, w_qkv, b_qkv, qb, kbf, vb);
    k_attn <<<(BB * NH * 16), 256, 0, stream>>>(qb, kbf, vb, aoT);
    k_proj <<<(BB * 16 * 128) / 8, 256, 0, stream>>>(aoT, w_out, b_out, x, out);
}
